// BatchedMonomialFactor_62302795596493
// MI455X (gfx1250) — matrix operand order repaired and checked by reading, not compiled, not run on hardware
//
#include <hip/hip_runtime.h>

// Fused kernel for BatchedMonomialFactor on gfx1250 (MI455X):
//   logits GEMM (bf16 WMMA) -> Sinkhorn (LDS) -> hard-perm scatter (ds_add_f32)
//   -> sigmoid/tanh gating -> output.
// Constants from the reference: R=64, B=16, TAU=0.5, 5 Sinkhorn iters, D=1024.

#define D_MODEL   1024
#define R_BLOCKS  64
#define B_DIM     16
#define TAU_INV   2.0f
#define SINK_IT   5

#define KCH 128          // K chunk staged per barrier round
#define NCH 288          // 256 perm cols + 16 diag + 16 alpha
#define WTS 136          // wt row stride in halves (128 + 8 pad; 272B, 16B-aligned)
#define XBS 1032         // xb row stride in halves (1024 + 8 pad; 2064B, 16B-aligned)

typedef __attribute__((ext_vector_type(16))) __bf16 v16bf;
typedef __attribute__((ext_vector_type(8)))  __bf16 v8bf;
typedef __attribute__((ext_vector_type(8)))  float  v8f;
typedef __attribute__((ext_vector_type(4)))  int    v4i;

typedef __attribute__((address_space(1))) v4i* gbl_v4i_p;
typedef __attribute__((address_space(3))) v4i* lds_v4i_p;

struct bfx2 { __bf16 x, y; };
struct bfx4 { __bf16 x, y, z, w; };
union  FragU { v16bf v; v8bf h[2]; };

// wt chunk buffer overlays the one-time f32 x staging buffer (async-to-LDS dest)
union SmemU {
    __bf16 wt[NCH * WTS];          // 78,336 B
    float  xstage[16 * D_MODEL];   // 65,536 B
};

// Inner MMA over one staged K chunk; NT = number of B tiles this wave owns.
// Loads A + all B fragments into distinct locals first so the scheduler can
// cover them with one s_wait_dscnt and issue the WMMAs back-to-back.
template<int NT>
__device__ __forceinline__ void chunk_mma(const __bf16* xb, const __bf16* wt,
                                          int aofs, const int* bofs, int kb,
                                          v8f* acc)
{
    #pragma unroll
    for (int ks = 0; ks < KCH / 32; ++ks) {
        const int kgA = kb * KCH + ks * 32;
        const int klB = ks * 32;
        FragU a;
        a.h[0] = *(const v8bf*)&xb[aofs + kgA];
        a.h[1] = *(const v8bf*)&xb[aofs + kgA + 8];
        FragU b[NT];
        #pragma unroll
        for (int t = 0; t < NT; ++t) {
            b[t].h[0] = *(const v8bf*)&wt[bofs[t] + klB];
            b[t].h[1] = *(const v8bf*)&wt[bofs[t] + klB + 8];
        }
        #pragma unroll
        for (int t = 0; t < NT; ++t)
            acc[t] = __builtin_amdgcn_wmma_f32_16x16x32_bf16(
                         false, a.v, false, b[t].v, (short)0, acc[t], false, false);
    }
}

__global__ __launch_bounds__(256, 1)
void monomial_factor_fused(const float* __restrict__ x,
                           const float* __restrict__ h,
                           const float* __restrict__ Wp,
                           const float* __restrict__ Wd,
                           const float* __restrict__ Wa,
                           float* __restrict__ out)
{
    __shared__ __bf16 xb[16 * XBS];        // x tile,  [m][k]  bf16
    __shared__ SmemU  u;                   // wt chunk  <-> f32 x staging
    __shared__ float  la[16 * 272];        // logits,  [m][i*17 + j]
    __shared__ float  rowlse[256];         // [m*16 + i]
    __shared__ float  hperm[256];          // [m*16 + i]
    __shared__ float  dpre[256];           // pre-tanh diag  [m*16 + i]
    __shared__ float  apre[256];           // pre-sigmoid    [m*16 + i]

    const int tid  = threadIdx.x;
    const int mb   = blockIdx.x;           // batch tile (16 rows)
    const int rblk = blockIdx.y;           // r block
    const int lane = tid & 31;
    const int wave = tid >> 5;
    const int hi   = lane >> 4;            // lane half (0/1)
    const int r16  = lane & 15;

    // ---- stage x tile (16 x 1024 f32) as bf16 in LDS ----
#if __has_builtin(__builtin_amdgcn_global_load_async_to_lds_b128)
    {
        // gfx1250 async copy: global f32 tile -> LDS staging (no VGPR roundtrip)
        const float* xbase = x + (size_t)mb * 16 * D_MODEL;
        #pragma unroll
        for (int it = 0; it < 16; ++it) {
            int f16i = it * 256 + tid;                       // 0..4095 16B chunks
            __builtin_amdgcn_global_load_async_to_lds_b128(
                (gbl_v4i_p)(xbase + f16i * 4),
                (lds_v4i_p)&u.xstage[f16i * 4],
                0, 0);
        }
#if __has_builtin(__builtin_amdgcn_s_wait_asynccnt)
        __builtin_amdgcn_s_wait_asynccnt(0);
#else
        asm volatile("s_wait_asynccnt 0x0" ::: "memory");
#endif
        __syncthreads();
        // convert staged f32 -> bf16 xb
        const float4* xs = (const float4*)u.xstage;
        #pragma unroll
        for (int it = 0; it < 16; ++it) {
            int f4 = it * 256 + tid;
            int m  = f4 >> 8;
            int k4 = f4 & 255;
            float4 v = xs[f4];
            bfx4 p; p.x = (__bf16)v.x; p.y = (__bf16)v.y;
                    p.z = (__bf16)v.z; p.w = (__bf16)v.w;
            *(bfx4*)&xb[m * XBS + k4 * 4] = p;
        }
        __syncthreads();   // xstage free for reuse as wt
    }
#else
    {
        const float4* xs = (const float4*)(x + (size_t)mb * 16 * D_MODEL);
        #pragma unroll
        for (int it = 0; it < 16; ++it) {
            int f4 = it * 256 + tid;             // 0..4095 float4s
            int m  = f4 >> 8;                    // 256 float4 per row
            int k4 = f4 & 255;
            float4 v = xs[m * 256 + k4];
            bfx4 p; p.x = (__bf16)v.x; p.y = (__bf16)v.y;
                    p.z = (__bf16)v.z; p.w = (__bf16)v.w;
            *(bfx4*)&xb[m * XBS + k4 * 4] = p;
        }
    }
#endif

    v8f acc[3]; acc[0] = v8f{}; acc[1] = v8f{}; acc[2] = v8f{};
    const int n0 = wave * 2;
    const int n1 = n0 + 1;
    // A fragment: lane<16 -> k = e (0..15); lane>=16 -> k = 16 + e   (matches B)
    const int aofs = r16 * XBS + 16 * hi;
    // B fragment: lane<16 -> k = e (0..15); lane>=16 -> k = 16 + e   (same k order as A)
    int bofs[3];
    bofs[0] = (n0 * B_DIM + r16) * WTS + 16 * hi;
    bofs[1] = (n1 * B_DIM + r16) * WTS + 16 * hi;
    bofs[2] = ((256 + wave * B_DIM) + r16) * WTS + 16 * hi;  // diag (wave0) / alpha (wave1)

    for (int kb = 0; kb < D_MODEL / KCH; ++kb) {
        // ---- cooperatively stage wt[n][k] for this K chunk (f32 -> bf16, transposed) ----
        #pragma unroll 4
        for (int itl = 0; itl < (KCH / 2) * NCH / 256; ++itl) {   // 72 iters
            int flat = itl * 256 + tid;          // 0..18431
            int kp = flat / NCH;                 // k pair 0..63
            int c  = flat - kp * NCH;            // column 0..287
            size_t kg = (size_t)(kb * KCH + 2 * kp);
            float f0, f1;
            if (c < 256) {
                const float* p = Wp + kg * (R_BLOCKS * 256) + rblk * 256 + c;
                f0 = p[0]; f1 = p[R_BLOCKS * 256];
            } else if (c < 272) {
                const float* p = Wd + kg * (R_BLOCKS * 16) + rblk * 16 + (c - 256);
                f0 = p[0]; f1 = p[R_BLOCKS * 16];
            } else {
                const float* p = Wa + kg * (R_BLOCKS * 16) + rblk * 16 + (c - 272);
                f0 = p[0]; f1 = p[R_BLOCKS * 16];
            }
            bfx2 pk; pk.x = (__bf16)f0; pk.y = (__bf16)f1;
            *(bfx2*)&u.wt[c * WTS + 2 * kp] = pk;
        }
        if (kb + 1 < D_MODEL / KCH) {
            // warm L2 for the next W_perm chunk (global_prefetch_b8)
            const float* np = Wp + (size_t)((kb + 1) * KCH + (tid >> 2)) * (R_BLOCKS * 256)
                                 + rblk * 256 + (tid & 3) * 64;
            __builtin_prefetch(np, 0, 1);
        }
        __syncthreads();

        // ---- 4 WMMA K-steps of 32 over this chunk (wave-uniform tile-count split) ----
        if (wave < 2) chunk_mma<3>(xb, u.wt, aofs, bofs, kb, acc);
        else          chunk_mma<2>(xb, u.wt, aofs, bofs, kb, acc);
        __syncthreads();
    }

    // ---- spill C tiles: logits (x 1/tau) + diag/alpha pre-activations ----
    #pragma unroll
    for (int v = 0; v < 8; ++v) {
        int m = v + 8 * hi;                       // ISA 32-bit C/D layout
        la[m * 272 + n0 * 17 + r16] = acc[0][v] * TAU_INV;
        la[m * 272 + n1 * 17 + r16] = acc[1][v] * TAU_INV;
    }
    if (wave == 0) {
        #pragma unroll
        for (int v = 0; v < 8; ++v) dpre[(v + 8 * hi) * 16 + r16] = acc[2][v];
    } else if (wave == 1) {
        #pragma unroll
        for (int v = 0; v < 8; ++v) apre[(v + 8 * hi) * 16 + r16] = acc[2][v];
    }
    hperm[tid] = 0.0f;

    const int sm = tid >> 4;                      // batch row within tile
    const int sj = tid & 15;                      // column j (phase2) / row i (phase1)
    const float hval = h[(size_t)(mb * 16 + sm) * (R_BLOCKS * B_DIM) + rblk * B_DIM + sj];
    __syncthreads();

    // ---- Sinkhorn: 5 x (row-LSE over j, then col-LSE over i), fused argmax+scatter ----
    for (int it = 0; it < SINK_IT; ++it) {
        {   // phase 1: thread = (m=sm, i=sj), logsumexp over j
            const float* row = &la[sm * 272 + sj * 17];
            float mx = row[0];
            #pragma unroll
            for (int j = 1; j < 16; ++j) mx = fmaxf(mx, row[j]);
            float s = 0.0f;
            #pragma unroll
            for (int j = 0; j < 16; ++j) s += __expf(row[j] - mx);
            rowlse[tid] = mx + __logf(s);
        }
        __syncthreads();
        // phase 2: thread = (m=sm, j=sj), column held in registers
        float v[16];
        float mx = -3.402823466e38f;
        #pragma unroll
        for (int i = 0; i < 16; ++i) {
            v[i] = la[sm * 272 + i * 17 + sj] - rowlse[sm * 16 + i];
            mx = fmaxf(mx, v[i]);
        }
        float s = 0.0f;
        #pragma unroll
        for (int i = 0; i < 16; ++i) s += __expf(v[i] - mx);
        float cl = mx + __logf(s);
        if (it < SINK_IT - 1) {
            #pragma unroll
            for (int i = 0; i < 16; ++i)
                la[sm * 272 + i * 17 + sj] = v[i] - cl;
        } else {
            // argmax over i (first max, like jnp.argmax); forward perm == hard
            int idx = 0; float best = v[0];
            #pragma unroll
            for (int i = 1; i < 16; ++i) { if (v[i] > best) { best = v[i]; idx = i; } }
            float sa = __expf(best - cl);                 // soft value at argmax
            float contrib = ((1.0f - sa) + sa) * hval;    // matches (hard - soft) + soft in fp
            atomicAdd(&hperm[sm * 16 + idx], contrib);    // ds_add_f32
        }
        __syncthreads();
    }

    // ---- gate and write output: thread = (m=sm, i=sj) ----
    float dv = tanhf(dpre[tid]);
    float av = 1.0f / (1.0f + __expf(-apre[tid]));
    out[(size_t)(mb * 16 + sm) * (R_BLOCKS * B_DIM) + rblk * B_DIM + sj] =
        av * dv * hperm[tid];
}

extern "C" void kernel_launch(void* const* d_in, const int* in_sizes, int n_in,
                              void* d_out, int out_size, void* d_ws, size_t ws_size,
                              hipStream_t stream) {
    const float* x  = (const float*)d_in[0];   // (batch, 1024)
    const float* h  = (const float*)d_in[1];   // (batch, 1024)
    const float* Wp = (const float*)d_in[2];   // (1024, 16384)
    const float* Wd = (const float*)d_in[3];   // (1024, 1024)
    const float* Wa = (const float*)d_in[4];   // (1024, 1024)
    float* out = (float*)d_out;

    const int batch = in_sizes[0] / D_MODEL;
    dim3 grid(batch / 16, R_BLOCKS);
    monomial_factor_fused<<<grid, 256, 0, stream>>>(x, h, Wp, Wd, Wa, out);
}
